// HNet_88158498717863
// MI455X (gfx1250) — compile-verified, matrix operand order checked by reading, not run on hardware
//
#include <hip/hip_runtime.h>
#include <math.h>

// ---------------- constants (match reference) ----------------
#define VOCAB 50257
#define BB    4
#define TT    256
#define D1_   64
#define D2_   128
#define D3_   256
#define K1_   64
#define K2_   16
#define SDIM_ 10
#define NH_   4
#define SKIP_ 1

// ---------------- WMMA path selection ----------------
#if defined(__AMDGCN__)
#  if __has_builtin(__builtin_amdgcn_wmma_f32_16x16x4_f32)
#    define WMMA_PATH 1
#  elif __has_builtin(__builtin_amdgcn_wmma_f32_16x16x32_bf16)
#    define WMMA_PATH 2
#  else
#    define WMMA_PATH 0
#  endif
#else
#  define WMMA_PATH 0
#endif

typedef __attribute__((ext_vector_type(2))) float v2f;
typedef __attribute__((ext_vector_type(8))) float v8f;
#if WMMA_PATH == 2
typedef __attribute__((ext_vector_type(16))) __bf16 v16bf;
__device__ inline int k16map(int e, int half) {
  int v = e >> 1, p = e & 1;
  int base = (v < 4) ? (2 * v) : (16 + 2 * (v - 4));
  return base + 8 * half + p;
}
#endif

__device__ __host__ inline int imin(int a, int b) { return a < b ? a : b; }

__device__ inline float wave_sum(float v) {
  for (int m = 16; m >= 1; m >>= 1) v += __shfl_xor(v, m, 32);
  return v;
}
__device__ inline float wave_max(float v) {
  for (int m = 16; m >= 1; m >>= 1) v = fmaxf(v, __shfl_xor(v, m, 32));
  return v;
}
__device__ inline float gelu_tanh(float x) {
  const float c = 0.7978845608028654f;
  float t = tanhf(c * (x + 0.044715f * x * x * x));
  return 0.5f * x * (1.0f + t);
}

// ---------------- GEMM: D = epilogue(A(MxK) @ B(KxN)) ----------------
// act: 0 = none, 1 = gelu(tanh).  v = act(acc + bias[col]) + resid[row,col]
__global__ void __launch_bounds__(32)
gemm_wmma_k(const float* __restrict__ A, int lda,
            const float* __restrict__ Bm, int ldb,
            const float* __restrict__ bias,
            const float* __restrict__ resid,
            float* __restrict__ D, int ldd,
            int M, int N, int K, int act) {
  const int lane = threadIdx.x;
  const int half = lane >> 4;
  const int l16  = lane & 15;
  const int m0 = blockIdx.y << 4;
  const int n0 = blockIdx.x << 4;
  v8f c = {};
  const int ar = imin(m0 + l16, M - 1);
  const int bc = imin(n0 + l16, N - 1);
#if WMMA_PATH == 1
  for (int k = 0; k < K; k += 4) {
    v2f a, b;
    const int ka = k + 2 * half;
    a.x = A[(size_t)ar * lda + ka];
    a.y = A[(size_t)ar * lda + ka + 1];
    b.x = Bm[(size_t)ka * ldb + bc];
    b.y = Bm[(size_t)(ka + 1) * ldb + bc];
    c = __builtin_amdgcn_wmma_f32_16x16x4_f32(false, a, false, b, (short)0, c, false, false);
  }
#elif WMMA_PATH == 2
  for (int k0 = 0; k0 < K; k0 += 32) {
    v16bf a, b;
    for (int e = 0; e < 16; ++e) {
      int kk = k0 + k16map(e, half);
      float av = (kk < K) ? A[(size_t)ar * lda + kk] : 0.0f;
      float bv = (kk < K) ? Bm[(size_t)kk * ldb + bc] : 0.0f;
      a[e] = (__bf16)av;
      b[e] = (__bf16)bv;
    }
    c = __builtin_amdgcn_wmma_f32_16x16x32_bf16(false, a, false, b, (short)0, c, false, false);
  }
#else
  (void)A; (void)Bm; (void)lda; (void)ldb; (void)ar; (void)bc;
#endif
  for (int r = 0; r < 8; ++r) {
    const int row = m0 + r + 8 * half;
    const int col = n0 + l16;
    if (row < M && col < N) {
      float v = c[r];
      if (bias) v += bias[col];
      if (act == 1) v = gelu_tanh(v);
      if (resid) v += resid[(size_t)row * ldd + col];
      D[(size_t)row * ldd + col] = v;
    }
  }
}

// ---------------- fused attention (one wave per b,h,16-query tile) ----------------
__global__ void __launch_bounds__(32)
attn_k(const float* __restrict__ Q, int qStride, int qLen,
       const float* __restrict__ Kp, int kStride,
       const float* __restrict__ Vp, int vStride, int kLen,
       float* __restrict__ O, int oStride,
       int hd, float scale, int causal) {
  __shared__ float S[16][257];
  const int lane = threadIdx.x;
  const int half = lane >> 4;
  const int l16  = lane & 15;
  const int qt = blockIdx.x, h = blockIdx.y, b = blockIdx.z;
  const int q0 = qt << 4;
  const float* Qb = Q  + (size_t)b * qLen * qStride + h * hd;
  const float* Kb = Kp + (size_t)b * kLen * kStride + h * hd;
  const float* Vb = Vp + (size_t)b * kLen * vStride + h * hd;
  float*       Ob = O  + (size_t)b * qLen * oStride + h * hd;
  const int ktiles = (kLen + 15) >> 4;
  const int ktUsed = causal ? imin(ktiles, qt + 1) : ktiles;
  const int ncol = ktUsed << 4;
  const int qr = imin(q0 + l16, qLen - 1);

  // phase 1: S = scale * Q Kt  (+ masks)
  for (int kt = 0; kt < ktUsed; ++kt) {
    v8f c = {};
    const int kc = imin((kt << 4) + l16, kLen - 1);
#if WMMA_PATH == 1
    for (int d = 0; d < hd; d += 4) {
      v2f a, bv;
      const int da = d + 2 * half;
      a.x  = Qb[(size_t)qr * qStride + da];
      a.y  = Qb[(size_t)qr * qStride + da + 1];
      bv.x = Kb[(size_t)kc * kStride + da];
      bv.y = Kb[(size_t)kc * kStride + da + 1];
      c = __builtin_amdgcn_wmma_f32_16x16x4_f32(false, a, false, bv, (short)0, c, false, false);
    }
#elif WMMA_PATH == 2
    for (int d0 = 0; d0 < hd; d0 += 32) {
      v16bf a, bv;
      for (int e = 0; e < 16; ++e) {
        int dd = d0 + k16map(e, half);
        float av = (dd < hd) ? Qb[(size_t)qr * qStride + dd] : 0.0f;
        float bb = (dd < hd) ? Kb[(size_t)kc * kStride + dd] : 0.0f;
        a[e] = (__bf16)av;
        bv[e] = (__bf16)bb;
      }
      c = __builtin_amdgcn_wmma_f32_16x16x32_bf16(false, a, false, bv, (short)0, c, false, false);
    }
#endif
    for (int r = 0; r < 8; ++r) {
      const int row = r + 8 * half;
      const int q = q0 + row;
      const int kcol = (kt << 4) + l16;
      float v = c[r] * scale;
      if (kcol >= kLen) v = -1e30f;
      else if (causal && kcol > q) v = -1e9f;
      S[row][(kt << 4) + l16] = v;
    }
  }
  __syncthreads();

  // phase 2: row softmax
  for (int m = 0; m < 16; ++m) {
    if (q0 + m >= qLen) continue;
    float mx = -1e30f;
    for (int cc = lane; cc < ncol; cc += 32) mx = fmaxf(mx, S[m][cc]);
    mx = wave_max(mx);
    float s = 0.0f;
    for (int cc = lane; cc < ncol; cc += 32) s += expf(S[m][cc] - mx);
    s = wave_sum(s);
    const float inv = 1.0f / s;
    for (int cc = lane; cc < ncol; cc += 32) S[m][cc] = expf(S[m][cc] - mx) * inv;
  }
  __syncthreads();

  // phase 3: O = P @ V
  for (int nt = 0; nt < hd; nt += 16) {
    v8f c = {};
#if WMMA_PATH == 1
    for (int k = 0; k < ncol; k += 4) {
      v2f a, bv;
      const int kk = k + 2 * half;
      a.x = S[l16][kk];
      a.y = S[l16][kk + 1];
      const int kv0 = imin(kk,     kLen - 1);
      const int kv1 = imin(kk + 1, kLen - 1);
      bv.x = Vb[(size_t)kv0 * vStride + nt + l16];
      bv.y = Vb[(size_t)kv1 * vStride + nt + l16];
      c = __builtin_amdgcn_wmma_f32_16x16x4_f32(false, a, false, bv, (short)0, c, false, false);
    }
#elif WMMA_PATH == 2
    for (int k0 = 0; k0 < ncol; k0 += 32) {
      v16bf a, bv;
      for (int e = 0; e < 16; ++e) {
        int kk = k0 + k16map(e, half);
        float av = (kk < ncol) ? S[l16][kk] : 0.0f;
        int kv = imin(kk, kLen - 1);
        float bb = (kk < ncol) ? Vb[(size_t)kv * vStride + nt + l16] : 0.0f;
        a[e] = (__bf16)av;
        bv[e] = (__bf16)bb;
      }
      c = __builtin_amdgcn_wmma_f32_16x16x32_bf16(false, a, false, bv, (short)0, c, false, false);
    }
#endif
    for (int r = 0; r < 8; ++r) {
      const int row = q0 + r + 8 * half;
      if (row < qLen) Ob[(size_t)row * oStride + nt + l16] = c[r];
    }
  }
}

// ---------------- layernorm (wave per row) ----------------
__global__ void __launch_bounds__(32)
ln_k(const float* __restrict__ x, const float* __restrict__ w,
     float* __restrict__ y, int D) {
  const int row = blockIdx.x, lane = threadIdx.x;
  const float* xr = x + (size_t)row * D;
  float* yr = y + (size_t)row * D;
  float s = 0.0f;
  for (int i = lane; i < D; i += 32) s += xr[i];
  s = wave_sum(s);
  const float mu = s / (float)D;
  float v = 0.0f;
  for (int i = lane; i < D; i += 32) { float d = xr[i] - mu; v += d * d; }
  v = wave_sum(v) / (float)D;
  const float inv = rsqrtf(v + 1e-5f);
  for (int i = lane; i < D; i += 32) yr[i] = (xr[i] - mu) * inv * w[i];
}

// ---------------- row softmax (wave per row) ----------------
__global__ void __launch_bounds__(32)
softmax_rows_k(const float* __restrict__ x, float* __restrict__ y, int N) {
  const int row = blockIdx.x, lane = threadIdx.x;
  const float* xr = x + (size_t)row * N;
  float* yr = y + (size_t)row * N;
  float mx = -1e30f;
  for (int i = lane; i < N; i += 32) mx = fmaxf(mx, xr[i]);
  mx = wave_max(mx);
  float s = 0.0f;
  for (int i = lane; i < N; i += 32) s += expf(xr[i] - mx);
  s = wave_sum(s);
  const float inv = 1.0f / s;
  for (int i = lane; i < N; i += 32) yr[i] = expf(xr[i] - mx) * inv;
}

// ---------------- gate MLP: relu(x@g1w+g1b)@g2w+g2b (wave per row) ----------------
__global__ void __launch_bounds__(32)
gate_mlp_k(const float* __restrict__ scan,
           const float* __restrict__ g1w, const float* __restrict__ g1b,
           const float* __restrict__ g2w, const float* __restrict__ g2b,
           float* __restrict__ out, int od) {
  const int row = blockIdx.x, lane = threadIdx.x;
  const float* xr = scan + (size_t)row * od;
  const int H = od >> 2;
  float acc = 0.0f;
  for (int j = lane; j < H; j += 32) {
    float s = g1b[j];
    for (int d = 0; d < od; ++d) s = fmaf(xr[d], g1w[d * H + j], s);
    acc += fmaxf(s, 0.0f) * g2w[j];
  }
  acc = wave_sum(acc);
  if (lane == 0) out[row] = acc + g2b[0];
}

// ---------------- top-k gather (stable, deterministic) ----------------
__global__ void __launch_bounds__(256)
topk_gather_k(const float* __restrict__ gate, const float* __restrict__ x,
              float* __restrict__ out, int N, int k, int C) {
  __shared__ float g[256];
  __shared__ int sel[256];
  const int b = blockIdx.x, i = threadIdx.x;
  g[i] = (i < N) ? gate[(size_t)b * N + i] : -1e30f;
  __syncthreads();
  int sv = 0;
  if (i < N) {
    const float gi = g[i];
    int cnt = 0;
    for (int j = 0; j < N; ++j) {
      const float gj = g[j];
      cnt += (gj > gi) || (gj == gi && j < i);
    }
    sv = (cnt < k);
  }
  sel[i] = sv;
  __syncthreads();
  if (i < N && sv) {
    int pos = 0;
    for (int j = 0; j < i; ++j) pos += sel[j];
    const float gi = g[i];
    const float* src = x + ((size_t)b * N + i) * C;
    float* dst = out + ((size_t)b * k + pos) * C;
    for (int c = 0; c < C; ++c) dst[c] = src[c] * gi;
  }
}

// ---------------- top-k scatter + positional add ----------------
__global__ void __launch_bounds__(256)
scatter_pos_k(const float* __restrict__ sg, const float* __restrict__ x,
              const float* __restrict__ posw, const float* __restrict__ posb,
              float* __restrict__ out, int oseq, int iT, int C) {
  __shared__ float g[256];
  __shared__ int sel[256];
  const int b = blockIdx.x, i = threadIdx.x;
  g[i] = (i < oseq) ? sg[(size_t)b * oseq + i] : -1e30f;
  __syncthreads();
  int sv = 0;
  if (i < oseq) {
    const float gi = g[i];
    int cnt = 0;
    for (int j = 0; j < oseq; ++j) {
      const float gj = g[j];
      cnt += (gj > gi) || (gj == gi && j < i);
    }
    sv = (cnt < iT);
  }
  sel[i] = sv;
  __syncthreads();
  if (i < oseq) {
    float* dst = out + ((size_t)b * oseq + i) * C;
    if (sv) {
      int pos = 0;
      for (int j = 0; j < i; ++j) pos += sel[j];
      const float gi = g[i];
      const float* src = x + ((size_t)b * iT + pos) * C;
      for (int c = 0; c < C; ++c) dst[c] = src[c] * gi + posw[(size_t)i * C + c] + posb[c];
    } else {
      for (int c = 0; c < C; ++c) dst[c] = posw[(size_t)i * C + c] + posb[c];
    }
  }
}

// ---------------- misc elementwise ----------------
__global__ void embed_k(const int* __restrict__ idx, const float* __restrict__ tok,
                        const float* __restrict__ pos, float* __restrict__ out, int T, int D) {
  const int r = blockIdx.x, d = threadIdx.x;
  const int token = idx[r];
  const int t = r % T;
  out[(size_t)r * D + d] = tok[(size_t)token * D + d] + pos[(size_t)t * D + d];
}

__global__ void pack_sdim_k(const float* __restrict__ x, float* __restrict__ out,
                            int rows, int C) {
  const int i = blockIdx.x * blockDim.x + threadIdx.x;
  const int total = rows * SDIM_;
  if (i < total) {
    const int r = i / SDIM_, s = i % SDIM_;
    out[i] = x[(size_t)r * C + s];
  }
}

__global__ void shift_copy_k(const float* __restrict__ src, float* __restrict__ dst,
                             int Tsrc, int Tdst, int C, int off) {
  const int bt = blockIdx.x, c = threadIdx.x;
  const int b = bt / Tdst, t = bt % Tdst;
  if (c < C)
    dst[((size_t)b * Tdst + t) * C + c] = src[((size_t)b * Tsrc + t + off) * C + c];
}

// ---------------- loss reductions (deterministic, no atomics) ----------------
__global__ void __launch_bounds__(256)
mse_partial_k(const float* __restrict__ a, const float* __restrict__ b,
              float* __restrict__ part, int n) {
  __shared__ float red[256];
  const int tid = threadIdx.x;
  float s = 0.0f;
  for (int i = blockIdx.x * 256 + tid; i < n; i += 64 * 256) {
    const float d = a[i] - b[i];
    s = fmaf(d, d, s);
  }
  red[tid] = s;
  __syncthreads();
  for (int st = 128; st > 0; st >>= 1) {
    if (tid < st) red[tid] += red[tid + st];
    __syncthreads();
  }
  if (tid == 0) part[blockIdx.x] = red[0];
}

__global__ void __launch_bounds__(256)
ce_row_k(const float* __restrict__ logits, const int* __restrict__ tgt,
         float* __restrict__ rowloss, int Vv) {
  __shared__ float red[256];
  const int row = blockIdx.x, tid = threadIdx.x;
  const float* base = logits + (size_t)row * Vv;
  float mx = -1e30f;
  for (int i = tid; i < Vv; i += 256) mx = fmaxf(mx, base[i]);
  red[tid] = mx;
  __syncthreads();
  for (int s = 128; s > 0; s >>= 1) {
    if (tid < s) red[tid] = fmaxf(red[tid], red[tid + s]);
    __syncthreads();
  }
  mx = red[0];
  __syncthreads();
  float sum = 0.0f;
  for (int i = tid; i < Vv; i += 256) sum += expf(base[i] - mx);
  red[tid] = sum;
  __syncthreads();
  for (int s = 128; s > 0; s >>= 1) {
    if (tid < s) red[tid] += red[tid + s];
    __syncthreads();
  }
  if (tid == 0) rowloss[row] = (logf(red[0]) + mx) - base[tgt[row]];
}

__global__ void final_loss_k(const float* __restrict__ rowloss, int nrows,
                             const float* __restrict__ p0, float inv0,
                             const float* __restrict__ p1, float inv1,
                             float* __restrict__ out) {
  if (threadIdx.x == 0 && blockIdx.x == 0) {
    float ce = 0.0f;
    for (int i = 0; i < nrows; ++i) ce += rowloss[i];
    ce /= (float)nrows;
    float c0 = 0.0f, c1 = 0.0f;
    for (int i = 0; i < 64; ++i) { c0 += p0[i]; c1 += p1[i]; }
    out[0] = ce + c0 * inv0 + c1 * inv1;
  }
}

// =================== host-side orchestration ===================
namespace {

struct Arena {
  char* base;
  size_t off;
  float* f(size_t n) {
    size_t o = off;
    off += ((n * sizeof(float) + 255) / 256) * 256;
    return (float*)(base + o);
  }
};

struct BlockP { const float *ln1, *wqkv, *cproj, *ln2, *fc, *proj; };
struct SBlockP { const float *ln1, *ln2, *ln3, *wq, *wk, *wv, *cproj, *fc, *proj; };
struct LayerP {
  BlockP up;
  const float *upw, *upb, *posw, *posb;
  SBlockP ds;
  const float *dpw, *dpb;
  BlockP down;
  const float *g1w, *g1b, *g2w, *g2b, *dgw, *dgb;
  int od, idim, oseq, iseq;
};

inline void gemm(hipStream_t st, const float* A, int lda, const float* B, int ldb,
                 const float* bias, const float* resid, float* D, int ldd,
                 int M, int N, int K, int act) {
  dim3 g((N + 15) / 16, (M + 15) / 16);
  gemm_wmma_k<<<g, 32, 0, st>>>(A, lda, B, ldb, bias, resid, D, ldd, M, N, K, act);
}
inline void layernorm(hipStream_t st, const float* x, const float* w, float* y, int rows, int D) {
  ln_k<<<rows, 32, 0, st>>>(x, w, y, D);
}
inline void softmax_rows(hipStream_t st, const float* x, float* y, int rows, int N) {
  softmax_rows_k<<<rows, 32, 0, st>>>(x, y, N);
}
inline void attention(hipStream_t st, const float* Q, int qs, int qLen,
                      const float* K, int ks, const float* V, int vs, int kLen,
                      float* O, int os, int Bb, int hd, float scale, int causal) {
  dim3 g((qLen + 15) / 16, NH_, Bb);
  attn_k<<<g, 32, 0, st>>>(Q, qs, qLen, K, ks, V, vs, kLen, O, os, hd, scale, causal);
}

void run_block(hipStream_t st, Arena& ar, const float* x, float* out,
               int Bb, int Tt, int C, const BlockP& p) {
  const int M = Bb * Tt;
  float* tln = ar.f((size_t)M * C);
  float* qkv = ar.f((size_t)M * 3 * C);
  float* att = ar.f((size_t)M * C);
  float* x1  = ar.f((size_t)M * C);
  float* tl2 = ar.f((size_t)M * C);
  float* h   = ar.f((size_t)M * 4 * C);
  layernorm(st, x, p.ln1, tln, M, C);
  gemm(st, tln, C, p.wqkv, 3 * C, nullptr, nullptr, qkv, 3 * C, M, 3 * C, C, 0);
  const int hd = C / NH_;
  attention(st, qkv, 3 * C, Tt, qkv + C, 3 * C, qkv + 2 * C, 3 * C, Tt,
            att, C, Bb, hd, 1.0f / sqrtf((float)hd), 1);
  gemm(st, att, C, p.cproj, C, nullptr, x, x1, C, M, C, C, 0);
  layernorm(st, x1, p.ln2, tl2, M, C);
  gemm(st, tl2, C, p.fc, 4 * C, nullptr, nullptr, h, 4 * C, M, 4 * C, C, 1);
  gemm(st, h, 4 * C, p.proj, C, nullptr, x1, out, C, M, C, 4 * C, 0);
}

void run_sblock(hipStream_t st, Arena& ar, const float* x, const float* sx, float* out,
                int Bb, int iT, int oT, int C, const SBlockP& p) {
  const int Mi = Bb * iT, Mo = Bb * oT;
  float* lnx  = ar.f((size_t)Mi * C);
  float* lnsx = ar.f((size_t)Mo * C);
  float* q    = ar.f((size_t)Mo * C);
  float* kk   = ar.f((size_t)Mi * C);
  float* vv   = ar.f((size_t)Mi * C);
  float* catt = ar.f((size_t)Mo * C);
  float* sx1  = ar.f((size_t)Mo * C);
  float* tl   = ar.f((size_t)Mo * C);
  float* h    = ar.f((size_t)Mo * 4 * C);
  layernorm(st, x, p.ln1, lnx, Mi, C);
  layernorm(st, sx, p.ln2, lnsx, Mo, C);
  gemm(st, lnsx, C, p.wq, C, nullptr, nullptr, q, C, Mo, C, C, 0);
  gemm(st, lnx, C, p.wk, C, nullptr, nullptr, kk, C, Mi, C, C, 0);
  gemm(st, lnx, C, p.wv, C, nullptr, nullptr, vv, C, Mi, C, C, 0);
  attention(st, q, C, oT, kk, C, vv, C, iT, catt, C, Bb, C / NH_, 1.0f, 0);
  gemm(st, catt, C, p.cproj, C, nullptr, sx, sx1, C, Mo, C, C, 0);
  layernorm(st, sx1, p.ln3, tl, Mo, C);
  gemm(st, tl, C, p.fc, 4 * C, nullptr, nullptr, h, 4 * C, Mo, 4 * C, C, 1);
  gemm(st, h, 4 * C, p.proj, C, nullptr, sx1, out, C, Mo, C, 4 * C, 0);
}

float* run_up(hipStream_t st, Arena& ar, const float* x, int Bb, int To, const LayerP& L) {
  const int od = L.od, k = L.iseq, idim = L.idim;
  const int M = Bb * To;
  float* scan = ar.f((size_t)M * od);
  run_block(st, ar, x, scan, Bb, To, od, L.up);
  float* gate = ar.f((size_t)M);
  gate_mlp_k<<<M, 32, 0, st>>>(scan, L.g1w, L.g1b, L.g2w, L.g2b, gate, od);
  float* gateS = ar.f((size_t)M);
  softmax_rows(st, gate, gateS, Bb, To);
  float* ssm = ar.f((size_t)M * od);
  softmax_rows(st, scan, ssm, M, od);
  float* gath = ar.f((size_t)Bb * k * od);
  topk_gather_k<<<Bb, 256, 0, st>>>(gateS, ssm, gath, To, k, od);
  float* out = ar.f((size_t)Bb * k * idim);
  gemm(st, gath, od, L.upw, idim, L.upb, nullptr, out, idim, Bb * k, idim, od, 0);
  return out;
}

float* run_down(hipStream_t st, Arena& ar, const float* x, const float* residual,
                int Bb, const LayerP& L) {
  const int iT = L.iseq, oT = L.oseq, idim = L.idim, od = L.od;
  float* pack = ar.f((size_t)Bb * iT * SDIM_);
  {
    const int total = Bb * iT * SDIM_;
    pack_sdim_k<<<(total + 255) / 256, 256, 0, st>>>(x, pack, Bb * iT, idim);
  }
  float* sgl = ar.f((size_t)Bb * oT);
  gemm(st, pack, iT * SDIM_, L.dgw, oT, L.dgb, nullptr, sgl, oT, Bb, oT, iT * SDIM_, 0);
  float* sg = ar.f((size_t)Bb * oT);
  softmax_rows(st, sgl, sg, Bb, oT);
  float* scat = ar.f((size_t)Bb * oT * idim);
  scatter_pos_k<<<Bb, 256, 0, st>>>(sg, x, L.posw, L.posb, scat, oT, iT, idim);
  float* pds = ar.f((size_t)Bb * oT * idim);
  run_sblock(st, ar, x, scat, pds, Bb, iT, oT, idim, L.ds);
  float* pd = ar.f((size_t)Bb * oT * od);
  gemm(st, pds, idim, L.dpw, od, L.dpb, residual, pd, od, Bb * oT, od, idim, 0);
  float* out = ar.f((size_t)Bb * oT * od);
  run_block(st, ar, pd, out, Bb, oT, od, L.down);
  return out;
}

} // namespace

extern "C" void kernel_launch(void* const* d_in, const int* in_sizes, int n_in,
                              void* d_out, int out_size, void* d_ws, size_t ws_size,
                              hipStream_t stream) {
  (void)in_sizes; (void)n_in; (void)out_size; (void)ws_size;
  int p = 0;
  auto F = [&]() { return (const float*)d_in[p++]; };
  auto takeBlock = [&]() {
    BlockP b;
    b.ln1 = F(); b.wqkv = F(); b.cproj = F(); b.ln2 = F(); b.fc = F(); b.proj = F();
    return b;
  };
  auto takeSBlock = [&]() {
    SBlockP s;
    s.ln1 = F(); s.ln2 = F(); s.ln3 = F(); s.wq = F(); s.wk = F(); s.wv = F();
    s.cproj = F(); s.fc = F(); s.proj = F();
    return s;
  };

  const int* idx = (const int*)d_in[p++];
  const int* tgt = (const int*)d_in[p++];
  const float* tok_emb = F();
  const float* pos_emb = F();
  const float* headw = F();
  const float* headb = F();

  LayerP L[2];
  const int ods[2]  = {D1_, D2_};
  const int idims[2] = {D2_, D3_};
  const int oseqs[2] = {TT, K1_};
  const int iseqs[2] = {K1_, K2_};
  for (int l = 0; l < 2; ++l) {
    L[l].up = takeBlock();
    L[l].upw = F(); L[l].upb = F(); L[l].posw = F(); L[l].posb = F();
    L[l].ds = takeSBlock();
    L[l].dpw = F(); L[l].dpb = F();
    L[l].down = takeBlock();
    L[l].g1w = F(); L[l].g1b = F(); L[l].g2w = F(); L[l].g2b = F();
    L[l].dgw = F(); L[l].dgb = F();
    L[l].od = ods[l]; L[l].idim = idims[l]; L[l].oseq = oseqs[l]; L[l].iseq = iseqs[l];
  }
  BlockP innerb  = takeBlock();
  BlockP innerb2 = takeBlock();

  Arena ar{(char*)d_ws, 0};
  hipStream_t st = stream;

  // ---- embedding ----
  float* x0 = ar.f((size_t)BB * TT * D1_);
  embed_k<<<BB * TT, D1_, 0, st>>>(idx, tok_emb, pos_emb, x0, TT, D1_);

  // ---- abstraction up ----
  float* x1 = run_up(st, ar, x0, BB, TT, L[0]);   // (B, K1, D2); residual = x0
  float* x2 = run_up(st, ar, x1, BB, K1_, L[1]);  // (B, K2, D3); residual = x1

  // ---- inner blocks ----
  float* x3 = ar.f((size_t)BB * K2_ * D3_);
  run_block(st, ar, x2, x3, BB, K2_, D3_, innerb);
  float* x4 = ar.f((size_t)BB * K2_ * D3_);
  run_block(st, ar, x3, x4, BB, K2_, D3_, innerb2);

  // ---- abstraction down ----
  float* x5 = run_down(st, ar, x4, x1, BB, L[1]); // (B, K1, D2)
  float* x6 = run_down(st, ar, x5, x0, BB, L[0]); // (B, T, D1)

  // ---- contrastive loss path ----
  const int Tc = TT - SKIP_;
  float* ci = ar.f((size_t)BB * Tc * D1_);
  float* co = ar.f((size_t)BB * Tc * D1_);
  shift_copy_k<<<BB * Tc, D1_, 0, st>>>(x0, ci, TT, Tc, D1_, SKIP_);
  shift_copy_k<<<BB * Tc, D1_, 0, st>>>(x6, co, TT, Tc, D1_, 0);

  float* ei0 = run_up(st, ar, ci, BB, Tc, L[0]);  // (B, K1, D2)
  float* eo0 = run_up(st, ar, co, BB, Tc, L[0]);
  float* cl0p = ar.f(64);
  mse_partial_k<<<64, 256, 0, st>>>(ei0, eo0, cl0p, BB * K1_ * D2_);

  float* ei1 = run_up(st, ar, ei0, BB, K1_, L[1]); // (B, K2, D3)
  float* eo1 = run_up(st, ar, eo0, BB, K1_, L[1]);
  float* cl1p = ar.f(64);
  mse_partial_k<<<64, 256, 0, st>>>(ei1, eo1, cl1p, BB * K2_ * D3_);

  // ---- head: logits straight into d_out ----
  float* logits = (float*)d_out;
  gemm(st, x6, D1_, headw, VOCAB, headb, nullptr, logits, VOCAB,
       BB * TT, VOCAB, D1_, 0);

  // ---- cross-entropy + final loss ----
  float* rowloss = ar.f((size_t)BB * TT);
  ce_row_k<<<BB * TT, 256, 0, st>>>(logits, tgt, rowloss, VOCAB);
  float* lossOut = logits + (size_t)BB * TT * VOCAB;
  final_loss_k<<<1, 32, 0, st>>>(rowloss, BB * TT,
                                 cl0p, 1.0f / (float)(BB * K1_ * D2_),
                                 cl1p, 1.0f / (float)(BB * K2_ * D3_),
                                 lossOut);
}
